// SymbolicPlanner_4131758539382
// MI455X (gfx1250) — matrix operand order repaired and checked by reading, not compiled, not run on hardware
//
#include <hip/hip_runtime.h>
#include <hip/hip_bf16.h>

// ---------------------------------------------------------------------------
// MoE MLP (3 experts, 362->1024->2048->2048->1024->362) + routed MSE loss.
// Route-first (3x FLOP cut vs dense reference); grouped GEMMs via
// V_WMMA_F32_16X16X32_BF16; register-prefetch software pipeline; packed
// b32 LDS stores for the transposed weight tile; fused loss epilogue with
// deterministic fixed-order final reduction.
// ---------------------------------------------------------------------------

typedef __attribute__((ext_vector_type(16))) __bf16 v16bf;
typedef __attribute__((ext_vector_type(8)))  __bf16 v8bf;
typedef __attribute__((ext_vector_type(8)))  float  v8f;

#define B_BATCH 8192
#define XDIM    362
#define NEXP    3

#define TM  128   // block tile M
#define TN  128   // block tile N
#define TK  32    // K step == WMMA K
#define LDT 40    // LDS row stride (elements, 80B: 16B-aligned, bank-spread)

__device__ __forceinline__ v16bf cat16(v8bf lo, v8bf hi) {
  v16bf r;
#pragma unroll
  for (int i = 0; i < 8; ++i) { r[i] = lo[i]; r[i + 8] = hi[i]; }
  return r;
}

// ---------------------------------------------------------------------------
// Grouped GEMM for expert e = blockIdx.z over rows [starts[e], +counts[e]):
//   out = act(Hin * W[e] + b[e])   (W f32 [E][Kreal][N], cvt->bf16 on stage)
// LOSS variant fuses (pred - next[perm[row]])^2 block reduction.
// ---------------------------------------------------------------------------
template <bool RELU, bool LOSS>
__global__ __launch_bounds__(256)
void moe_gemm(const __bf16* __restrict__ Hin, int ldin,
              const float* __restrict__ W, const float* __restrict__ bias,
              int Kreal, int N,
              __bf16* __restrict__ Hout, int ldout,
              const int* __restrict__ counts, const int* __restrict__ starts,
              const int* __restrict__ perm, const float* __restrict__ next,
              float* __restrict__ partials) {
  __shared__ __bf16 As[TM * LDT];
  __shared__ __bf16 Bs[TN * LDT];
  __shared__ float  red[256];

  const int tid = threadIdx.x;
  const int e   = blockIdx.z;
  const int cnt = counts[e];
  const int m0  = blockIdx.y * TM;
  const int bid = (blockIdx.z * gridDim.y + blockIdx.y) * gridDim.x + blockIdx.x;

  if (m0 >= cnt) {                       // padded grid: surplus blocks exit
    if (LOSS && tid == 0) partials[bid] = 0.f;
    return;
  }
  int mcount = cnt - m0; if (mcount > TM) mcount = TM;
  const int rowbase = starts[e] + m0;
  const int n0 = blockIdx.x * TN;
  const long long woff = (long long)e * Kreal * N;

  // wave coordinates: 8 waves as 2 (M) x 4 (N); each wave owns 64x32
  const int lane = tid & 31;
  const int w    = tid >> 5;
  const int wm   = w >> 2;        // 0..1
  const int wn   = w & 3;         // 0..3
  const int l16  = lane & 15;
  const int g    = lane >> 4;     // half-wave select per ISA WMMA layout

  v8f acc[4][2];
#pragma unroll
  for (int t = 0; t < 4; ++t)
#pragma unroll
    for (int u = 0; u < 2; ++u)
#pragma unroll
      for (int v = 0; v < 8; ++v) acc[t][u][v] = 0.f;

  // A staging: 2 threads/row, 16 bf16 each (two b128 load/store pairs)
  const int ar = tid >> 1;
  const int ac = (tid & 1) * 16;
  // B staging: each thread owns a k-pair (kb, kb+1) x 8 columns
  const int kb  = (tid & 15) * 2;       // 0..30, even
  const int ng8 = (tid >> 4) * 8;       // 0..120 step 8

  const int Kpad = ldin;                // multiple of TK

  // -------- register prefetch state --------
  v8bf  pa0, pa1;
  float wv0[8], wv1[8];

  auto ldA = [&](int k0) {
    v8bf z;
#pragma unroll
    for (int i = 0; i < 8; ++i) z[i] = (__bf16)0.f;
    pa0 = z; pa1 = z;
    if (ar < mcount) {
      const __bf16* src = Hin + (long long)(rowbase + ar) * ldin + k0 + ac;
      pa0 = *reinterpret_cast<const v8bf*>(src);
      pa1 = *reinterpret_cast<const v8bf*>(src + 8);
    }
  };
  auto ldB = [&](int k0) {
    const int kg0 = k0 + kb;
    const int kg1 = kg0 + 1;
    const float* r0 = W + woff + (long long)kg0 * N + n0 + ng8;
    const float* r1 = r0 + N;
#pragma unroll
    for (int j = 0; j < 4; ++j) {
      float2 v0; v0.x = 0.f; v0.y = 0.f;
      float2 v1; v1.x = 0.f; v1.y = 0.f;
      if ((n0 + ng8 + 2 * j) < N) {     // N even => pair fully in/out
        if (kg0 < Kreal) v0 = reinterpret_cast<const float2*>(r0)[j];
        if (kg1 < Kreal) v1 = reinterpret_cast<const float2*>(r1)[j];
      }
      wv0[2 * j] = v0.x; wv0[2 * j + 1] = v0.y;
      wv1[2 * j] = v1.x; wv1[2 * j + 1] = v1.y;
    }
  };
  auto stA = [&]() {
    *reinterpret_cast<v8bf*>(&As[ar * LDT + ac])     = pa0;
    *reinterpret_cast<v8bf*>(&As[ar * LDT + ac + 8]) = pa1;
  };
  auto stB = [&]() {                    // transposed [n][k], packed b32 stores
#pragma unroll
    for (int j = 0; j < 8; ++j) {
      union { unsigned u; __bf16 h[2]; } p;
      p.h[0] = (__bf16)wv0[j];          // k = kb     -> low half
      p.h[1] = (__bf16)wv1[j];          // k = kb + 1 -> high half
      *reinterpret_cast<unsigned*>(&Bs[(ng8 + j) * LDT + kb]) = p.u;
    }
  };

  // -------- software-pipelined main loop --------
  ldA(0); ldB(0);
  stA(); stB();
  __syncthreads();

  for (int k0 = 0; k0 < Kpad; k0 += TK) {
    const int  kn   = k0 + TK;
    const bool more = (kn < Kpad);
    if (more) { ldA(kn); ldB(kn); }     // prefetch next tile into VGPRs

    // fragments per ISA 16-bit A/B VGPR layouts + 8 WMMAs per wave
    v16bf afr[4], bfr[2];
#pragma unroll
    for (int t = 0; t < 4; ++t) {
      const __bf16* p = &As[(wm * 64 + t * 16 + l16) * LDT + g * 8];
      afr[t] = cat16(*reinterpret_cast<const v8bf*>(p),
                     *reinterpret_cast<const v8bf*>(p + 16));
    }
#pragma unroll
    for (int u = 0; u < 2; ++u) {
      const __bf16* p = &Bs[(wn * 32 + u * 16 + l16) * LDT + g * 8];
      bfr[u] = cat16(*reinterpret_cast<const v8bf*>(p),
                     *reinterpret_cast<const v8bf*>(p + 16));
    }
#pragma unroll
    for (int t = 0; t < 4; ++t)
#pragma unroll
      for (int u = 0; u < 2; ++u)
        acc[t][u] = __builtin_amdgcn_wmma_f32_16x16x32_bf16(
            false, afr[t], false, bfr[u], (short)0, acc[t][u], false, false);

    if (more) {
      __syncthreads();                  // all waves done reading this tile
      stA(); stB();                     // park prefetched regs into LDS
      __syncthreads();                  // tile visible for next iteration
    }
  }

  // ---------------- epilogue ----------------
  if (!LOSS) {
#pragma unroll
    for (int u = 0; u < 2; ++u) {
      const int ng = n0 + wn * 32 + u * 16 + l16;
      if (ng >= N) continue;
      const float bv = bias[e * N + ng];
#pragma unroll
      for (int t = 0; t < 4; ++t)
#pragma unroll
        for (int v = 0; v < 8; ++v) {
          const int ml = wm * 64 + t * 16 + g * 8 + v;  // C/D: M = v + 8*g
          if (ml < mcount) {
            float x = acc[t][u][v] + bv;
            if (RELU) x = fmaxf(x, 0.f);
            Hout[(long long)(rowbase + ml) * ldout + ng] = (__bf16)x;
          }
        }
    }
  } else {
    float ls = 0.f;
#pragma unroll
    for (int u = 0; u < 2; ++u) {
      const int ng = n0 + wn * 32 + u * 16 + l16;
      if (ng >= N) continue;
      const float bv = bias[e * N + ng];
#pragma unroll
      for (int t = 0; t < 4; ++t)
#pragma unroll
        for (int v = 0; v < 8; ++v) {
          const int ml = wm * 64 + t * 16 + g * 8 + v;
          if (ml < mcount) {
            const float x = acc[t][u][v] + bv;
            const int   i = perm[rowbase + ml];
            const float d = x - next[(long long)i * XDIM + ng];
            ls += d * d;
          }
        }
    }
    red[tid] = ls;
    __syncthreads();
#pragma unroll
    for (int s = 128; s > 0; s >>= 1) {
      if (tid < s) red[tid] += red[tid + s];
      __syncthreads();
    }
    if (tid == 0) partials[bid] = red[0];
  }
}

// ---------------------------------------------------------------------------
// Routing: deterministic partition of the batch by expert.
// ---------------------------------------------------------------------------
__global__ void k_init(int* counts) {
  if (threadIdx.x < NEXP) counts[threadIdx.x] = 0;
}

__global__ void k_count(const int* __restrict__ a, int* counts) {
  const int i = blockIdx.x * blockDim.x + threadIdx.x;
  atomicAdd(&counts[a[i]], 1);
}

__global__ void k_scan(const int* __restrict__ counts, int* starts) {
  starts[0] = 0;
  starts[1] = counts[0];
  starts[2] = counts[0] + counts[1];
}

// stable rank: perm[starts[e] + #{j<i : a[j]==a[i]}] = i  (order-deterministic)
__global__ void k_rank(const int* __restrict__ a, const int* __restrict__ starts,
                       int* __restrict__ perm) {
  __shared__ int ch[256];
  const int i = blockIdx.x * 256 + threadIdx.x;
  const int e = a[i];
  int rank = 0;
  for (int c0 = 0; c0 <= (int)blockIdx.x; ++c0) {
    __syncthreads();
    ch[threadIdx.x] = a[c0 * 256 + threadIdx.x];
    __syncthreads();
    const int lim = (c0 == (int)blockIdx.x) ? (int)threadIdx.x : 256;
    for (int j = 0; j < lim; ++j) rank += (ch[j] == e);
  }
  perm[starts[e] + rank] = i;
}

// gather x rows by perm, f32 -> bf16, pad K 362 -> 384 with zeros
__global__ void k_gather(const float* __restrict__ x, const int* __restrict__ perm,
                         __bf16* __restrict__ H0) {
  const int idx = blockIdx.x * blockDim.x + threadIdx.x;   // B*384
  const int row = idx / 384, c = idx % 384;
  const int i = perm[row];
  H0[(long long)row * 384 + c] =
      (c < XDIM) ? (__bf16)x[(long long)i * XDIM + c] : (__bf16)0.f;
}

// fixed-order sum of per-block partials -> mean
__global__ void k_final(const float* __restrict__ partials, int n, float* out) {
  float s = 0.f;
  for (int i = 0; i < n; ++i) s += partials[i];
  out[0] = s / (float)((long long)B_BATCH * XDIM);
}

// ---------------------------------------------------------------------------
extern "C" void kernel_launch(void* const* d_in, const int* in_sizes, int n_in,
                              void* d_out, int out_size, void* d_ws, size_t ws_size,
                              hipStream_t stream) {
  (void)in_sizes; (void)n_in; (void)out_size; (void)ws_size;
  const float* x   = (const float*)d_in[0];
  const float* nxt = (const float*)d_in[1];
  const int*   act = (const int*)d_in[2];
  const float* Wp[5] = {(const float*)d_in[3], (const float*)d_in[5],
                        (const float*)d_in[7], (const float*)d_in[9],
                        (const float*)d_in[11]};
  const float* bp[5] = {(const float*)d_in[4], (const float*)d_in[6],
                        (const float*)d_in[8], (const float*)d_in[10],
                        (const float*)d_in[12]};

  char*  ws       = (char*)d_ws;
  int*   counts   = (int*)(ws + 0);
  int*   starts   = (int*)(ws + 16);
  float* partials = (float*)(ws + 64);                 // 576 floats
  int*   perm     = (int*)(ws + 4096);                 // 8192 ints
  __bf16* HA      = (__bf16*)(ws + 65536);             // B x 2048 bf16
  __bf16* HB      = (__bf16*)(ws + 65536 + (size_t)B_BATCH * 2048 * 2);

  k_init  <<<1, 32, 0, stream>>>(counts);
  k_count <<<B_BATCH / 256, 256, 0, stream>>>(act, counts);
  k_scan  <<<1, 1, 0, stream>>>(counts, starts);
  k_rank  <<<B_BATCH / 256, 256, 0, stream>>>(act, starts, perm);
  k_gather<<<(B_BATCH * 384) / 256, 256, 0, stream>>>(x, perm, HA);

  const dim3 blk(256);
  // L0: K=362 (pad 384), N=1024 : HA(ld384) -> HB(ld1024), relu
  moe_gemm<true, false><<<dim3(1024 / TN, B_BATCH / TM, NEXP), blk, 0, stream>>>(
      HA, 384, Wp[0], bp[0], 362, 1024, HB, 1024, counts, starts,
      nullptr, nullptr, nullptr);
  // L1: K=1024, N=2048 : HB -> HA(ld2048), relu
  moe_gemm<true, false><<<dim3(2048 / TN, B_BATCH / TM, NEXP), blk, 0, stream>>>(
      HB, 1024, Wp[1], bp[1], 1024, 2048, HA, 2048, counts, starts,
      nullptr, nullptr, nullptr);
  // L2: K=2048, N=2048 : HA -> HB(ld2048), relu
  moe_gemm<true, false><<<dim3(2048 / TN, B_BATCH / TM, NEXP), blk, 0, stream>>>(
      HA, 2048, Wp[2], bp[2], 2048, 2048, HB, 2048, counts, starts,
      nullptr, nullptr, nullptr);
  // L3: K=2048, N=1024 : HB -> HA(ld1024), relu
  moe_gemm<true, false><<<dim3(1024 / TN, B_BATCH / TM, NEXP), blk, 0, stream>>>(
      HB, 2048, Wp[3], bp[3], 2048, 1024, HA, 1024, counts, starts,
      nullptr, nullptr, nullptr);
  // L4: K=1024, N=362 : HA -> fused squared-error partials (no relu)
  moe_gemm<false, true><<<dim3(3, B_BATCH / TM, NEXP), blk, 0, stream>>>(
      HA, 1024, Wp[4], bp[4], 1024, 362, nullptr, 0, counts, starts,
      perm, nxt, partials);

  k_final<<<1, 1, 0, stream>>>(partials, 3 * (B_BATCH / TM) * NEXP, (float*)d_out);
}
